// FFFLinear_27874337751164
// MI455X (gfx1250) — matrix operand order repaired and checked by reading, not compiled, not run on hardware
//
#include <hip/hip_runtime.h>
#include <hip/hip_bf16.h>
#include <stdint.h>

// ---------------- CDNA5 vector types ----------------
typedef __attribute__((ext_vector_type(16))) __bf16       v16bf;
typedef __attribute__((ext_vector_type(8)))  float        v8f;
typedef __attribute__((ext_vector_type(4)))  unsigned int u32x4;
typedef __attribute__((ext_vector_type(8)))  int          i32x8;
typedef __attribute__((ext_vector_type(4)))  int          i32x4;

#define D_IN        2048
#define NUM_LATENTS 65536
#define N_TOKENS    4096
#define K_TOP       32
#define TOPN        64            // 2k
#define TILE_L      256           // latents per outer iteration
#define KCHUNK      128           // K elements per TDM tile (bf16 -> 256B rows)
#define KSTEPS      (KCHUNK / 32)                    // 4 WMMA k-steps per chunk
#define CPLT        (D_IN / KCHUNK)                  // 16 chunks per latent tile
#define NCHUNKS     ((NUM_LATENTS / TILE_L) * CPLT)  // 4096
#define CAND_CAP    256           // == TILE_L worst case per token per tile
#define XPAD        8             // +16B per sX row: kills 16-way bank conflict
#define WPAD        8             // +16B per sW row via TDM pad: 272B row stride

union V16 { u32x4 q[2]; v16bf v; };

__device__ __forceinline__ unsigned short f2bf(float f) {
    unsigned u = __float_as_uint(f);
    unsigned r = u + 0x7FFFu + ((u >> 16) & 1u);     // round-to-nearest-even
    if ((u & 0x7F800000u) == 0x7F800000u) r = u;     // Inf/NaN passthrough
    return (unsigned short)(r >> 16);
}

// f32 -> bf16 bulk convert, 8 elements/thread
__global__ void f32_to_bf16_kernel(const float* __restrict__ src,
                                   unsigned short* __restrict__ dst,
                                   long long n) {
    long long i = ((long long)blockIdx.x * blockDim.x + threadIdx.x) * 8;
    if (i + 8 > n) return;
    float4 a = *(const float4*)(src + i);
    float4 b = *(const float4*)(src + i + 4);
    uint4 o;
    o.x = (unsigned)f2bf(a.x) | ((unsigned)f2bf(a.y) << 16);
    o.y = (unsigned)f2bf(a.z) | ((unsigned)f2bf(a.w) << 16);
    o.z = (unsigned)f2bf(b.x) | ((unsigned)f2bf(b.y) << 16);
    o.w = (unsigned)f2bf(b.z) | ((unsigned)f2bf(b.w) << 16);
    *(uint4*)(dst + i) = o;
}

// Tensor Data Mover: async 2D tile load (TILE_L rows x KCHUNK bf16, tensor row
// stride D_IN elems) from global W_bf16 into LDS, inserting 16B of LDS padding
// after each 256B row (pad_interval=5 -> 256B, pad_amount=3 -> 4 dwords) so
// B-operand row fetches spread across banks. Descriptor per ISA §8 (D#).
__device__ __forceinline__ void tdm_load_w_tile(const unsigned short* Wb,
                                                size_t elem_off,
                                                unsigned lds_off) {
    unsigned long long ga = (unsigned long long)(uintptr_t)(Wb + elem_off);
    u32x4 g0;
    g0[0] = 1u;                                   // count=1, user mode, no gather
    g0[1] = lds_off;                              // lds_addr (bytes)
    g0[2] = (unsigned)ga;                         // global_addr[31:0]
    g0[3] = ((unsigned)(ga >> 32) & 0x01FFFFFFu)  // global_addr[56:32]
            | (2u << 30);                         // type=2 ("image")
    i32x8 g1;
    g1[0] = (int)((1u << 16)                      // data_size=1 (2 bytes/elem)
            | (1u << 20)                          // pad_enable
            | (5u << 22)                          // pad_interval: 256B (per row)
            | (3u << 25));                        // pad_amount: 4 dwords (16B)
    g1[1] = (int)((unsigned)(D_IN & 0xFFFF) << 16);          // tensor_dim0 lo16
    g1[2] = (int)(((unsigned)D_IN >> 16)                     // tensor_dim0 hi16
            | (((unsigned)NUM_LATENTS & 0xFFFFu) << 16));    // tensor_dim1 lo16
    g1[3] = (int)(((unsigned)NUM_LATENTS >> 16)              // tensor_dim1 hi16
            | ((unsigned)KCHUNK << 16));                     // tile_dim0 = 128
    g1[4] = (int)TILE_L;                          // tile_dim1 = 256, tile_dim2 = 0
    g1[5] = (int)D_IN;                            // tensor_dim0_stride lo32
    g1[6] = 0;
    g1[7] = 0;
    i32x4 z4 = {0, 0, 0, 0};
#if __clang_major__ >= 23
    i32x8 z8 = {0, 0, 0, 0, 0, 0, 0, 0};
    __builtin_amdgcn_tensor_load_to_lds(g0, g1, z4, z4, z8, 0);
#else
    __builtin_amdgcn_tensor_load_to_lds(g0, g1, z4, z4, 0);
#endif
}

// 256 threads (8 wave32), 16 tokens/block, streams all 65536 latents.
__global__ __launch_bounds__(256) void fff_topk_kernel(
    const unsigned short* __restrict__ Wb,   // bf16 [NUM_LATENTS][D_IN]
    const unsigned short* __restrict__ Xb,   // bf16 [N_TOKENS][D_IN]
    const float* __restrict__ b_dec,         // [NUM_LATENTS]
    float* __restrict__ out_vals,            // [N_TOKENS][TOPN]
    int*   __restrict__ out_idx)             // [N_TOKENS][TOPN]
{
    __shared__ __align__(16) __bf16 sX[16][D_IN + XPAD];          // ~64.3 KB
    __shared__ __align__(16) __bf16 sW[2][TILE_L][KCHUNK + WPAD]; // 2 x 68 KB
    __shared__ float sTopV[16][TOPN];
    __shared__ int   sTopI[16][TOPN];
    __shared__ float sCandV[16][CAND_CAP];
    __shared__ int   sCandI[16][CAND_CAP];
    __shared__ int   sCcnt[16];
    __shared__ int   sTopCnt[16];
    __shared__ float sThr[16];

    const int tid  = threadIdx.x;
    const int lane = tid & 31;
    const int wv   = tid >> 5;           // 8 waves
    const int tok0 = blockIdx.x * 16;

    if (tid < 16) { sCcnt[tid] = 0; sTopCnt[tid] = 0; sThr[tid] = -3.4e38f; }

    // Prime the TDM pipeline with chunk 0 (wave 0 only; EXEC-independent DMA).
    if (tid < 32) tdm_load_w_tile(Wb, 0, (unsigned)(uintptr_t)&sW[0][0][0]);

    // Stage the x tile (bf16) into padded LDS while the TDM runs.
    for (int it = tid; it < 16 * (D_IN / 8); it += 256) {
        int row = it / (D_IN / 8);
        int c8  = it % (D_IN / 8);
        *(uint4*)&sX[row][c8 * 8] =
            *(const uint4*)(Xb + (size_t)(tok0 + row) * D_IN + (size_t)c8 * 8);
    }

    if (tid < 32) __builtin_amdgcn_s_wait_tensorcnt(0);
    __syncthreads();

    const int sel  = lane >> 4;          // half-wave select (ISA 16-bit layouts)
    const int trow = lane & 15;          // token row (A) / latent col (B)
    const int n0   = wv * 2;             // this wave's first 16-latent sub-tile

#pragma unroll 1
    for (int lt = 0; lt < NUM_LATENTS / TILE_L; ++lt) {
        v8f acc0 = {0.f, 0.f, 0.f, 0.f, 0.f, 0.f, 0.f, 0.f};
        v8f acc1 = {0.f, 0.f, 0.f, 0.f, 0.f, 0.f, 0.f, 0.f};
#pragma unroll 1
        for (int c = 0; c < CPLT; ++c) {
            const int G   = lt * CPLT + c;
            const int buf = G & 1;
            if (tid < 32 && (G + 1) < NCHUNKS) {      // prefetch next W tile
                const int Gn = G + 1;
                tdm_load_w_tile(Wb,
                                (size_t)(Gn >> 4) * TILE_L * D_IN +
                                (size_t)(Gn & 15) * KCHUNK,
                                (unsigned)(uintptr_t)&sW[buf ^ 1][0][0]);
            }
            // 8 WMMAs per barrier: 4 k-steps of 32, two 16-latent sub-tiles.
            // A (16x32 bf16): lanes 0-15 hold K 0..15, 16-31 hold K 16..31 contiguous
            // B (32x16 bf16): lane n holds column n, K 0..15 / 16..31 contiguous
            const int kbase = c * KCHUNK;
            const __bf16* xrow = &sX[trow][0];
            const __bf16* wr0  = &sW[buf][(n0 + 0) * 16 + trow][0];
            const __bf16* wr1  = &sW[buf][(n0 + 1) * 16 + trow][0];
#pragma unroll
            for (int ks = 0; ks < KSTEPS; ++ks) {
                const int ko = kbase + ks * 32;
                V16 A, B0, B1;
                A.q[0]  = *(const u32x4*)(xrow + ko + sel * 16);
                A.q[1]  = *(const u32x4*)(xrow + ko + sel * 16 + 8);
                const u32x4* w0 = (const u32x4*)(wr0 + ks * 32 + sel * 16);
                const u32x4* w1 = (const u32x4*)(wr1 + ks * 32 + sel * 16);
                B0.q[0] = w0[0]; B0.q[1] = w0[1];
                B1.q[0] = w1[0]; B1.q[1] = w1[1];
                acc0 = __builtin_amdgcn_wmma_f32_16x16x32_bf16(
                           false, A.v, false, B0.v, (short)0, acc0, false, false);
                acc1 = __builtin_amdgcn_wmma_f32_16x16x32_bf16(
                           false, A.v, false, B1.v, (short)0, acc1, false, false);
            }
            if (tid < 32) __builtin_amdgcn_s_wait_tensorcnt(0);
            __syncthreads();
        }

        // Append above-threshold scores: D[M=token][N=latent], VGPR r holds
        // M = r (lanes 0-15) / r+8 (lanes 16-31), N = lane&15.
        {
            const int L0 = lt * TILE_L + (n0 + 0) * 16 + trow;
            const int L1 = lt * TILE_L + (n0 + 1) * 16 + trow;
            const float bias0 = b_dec[L0];
            const float bias1 = b_dec[L1];
#pragma unroll
            for (int r = 0; r < 8; ++r) {
                const int t = r + (sel << 3);
                const float th = sThr[t];
                float v0 = acc0[r] + bias0;
                if (v0 > th) {
                    int p = atomicAdd(&sCcnt[t], 1);
                    if (p < CAND_CAP) { sCandV[t][p] = v0; sCandI[t][p] = L0; }
                }
                float v1 = acc1[r] + bias1;
                if (v1 > th) {
                    int p = atomicAdd(&sCcnt[t], 1);
                    if (p < CAND_CAP) { sCandV[t][p] = v1; sCandI[t][p] = L1; }
                }
            }
        }
        __syncthreads();

        // Merge candidates into per-token sorted top-64 (one thread per token).
        if (tid < 16) {
            const int t = tid;
            int nc = sCcnt[t]; if (nc > CAND_CAP) nc = CAND_CAP;
            int cnt = sTopCnt[t];
            for (int cdx = 0; cdx < nc; ++cdx) {
                const float v  = sCandV[t][cdx];
                const int   id = sCandI[t][cdx];
                if (cnt == TOPN && v <= sTopV[t][TOPN - 1]) continue;
                int j = (cnt < TOPN) ? cnt : (TOPN - 1);
                while (j > 0 && sTopV[t][j - 1] < v) {
                    sTopV[t][j] = sTopV[t][j - 1];
                    sTopI[t][j] = sTopI[t][j - 1];
                    --j;
                }
                sTopV[t][j] = v; sTopI[t][j] = id;
                if (cnt < TOPN) ++cnt;
            }
            sTopCnt[t] = cnt;
            sCcnt[t]   = 0;
            sThr[t]    = (cnt == TOPN) ? sTopV[t][TOPN - 1] : -3.4e38f;
        }
        __syncthreads();
    }

    // Emit: values = (v > values[k]) ? 1 : 0  (sig - stop_grad(sig) == 0), indices.
    for (int e = tid; e < 16 * TOPN; e += 256) {
        const int t = e >> 6;
        const int j = e & 63;
        const float thr = sTopV[t][K_TOP];
        const size_t o  = (size_t)(tok0 + t) * TOPN + j;
        out_vals[o] = (sTopV[t][j] > thr) ? 1.0f : 0.0f;
        out_idx[o]  = sTopI[t][j];
    }
}

extern "C" void kernel_launch(void* const* d_in, const int* in_sizes, int n_in,
                              void* d_out, int out_size, void* d_ws, size_t ws_size,
                              hipStream_t stream) {
    (void)in_sizes; (void)n_in; (void)out_size; (void)ws_size;
    const float* x = (const float*)d_in[0];
    const float* W = (const float*)d_in[1];
    const float* b = (const float*)d_in[2];
    // d_in[3] == k (scalar 32), compile-time constant here.

    unsigned short* Wb = (unsigned short*)d_ws;                      // 256 MiB
    unsigned short* Xb = Wb + (size_t)NUM_LATENTS * D_IN;            // +16 MiB

    {
        long long n = (long long)NUM_LATENTS * D_IN;
        f32_to_bf16_kernel<<<(int)(n / (256 * 8)), 256, 0, stream>>>(W, Wb, n);
    }
    {
        long long n = (long long)N_TOKENS * D_IN;
        f32_to_bf16_kernel<<<(int)(n / (256 * 8)), 256, 0, stream>>>(x, Xb, n);
    }

    float* outv = (float*)d_out;
    int*   outi = (int*)(outv + (size_t)N_TOKENS * TOPN);
    fff_topk_kernel<<<N_TOKENS / 16, 256, 0, stream>>>(Wb, Xb, b, outv, outi);
}
